// Attention_24704651887034
// MI455X (gfx1250) — matrix operand order repaired and checked by reading, not compiled, not run on hardware
//
#include <hip/hip_runtime.h>
#include <stdint.h>

// ---------------------------------------------------------------------------
// Types & helpers
// ---------------------------------------------------------------------------
typedef __attribute__((ext_vector_type(16))) __bf16 v16bf;
typedef __attribute__((ext_vector_type(8)))  float  v8f;
typedef __attribute__((ext_vector_type(4)))  int    v4i_t;
typedef __attribute__((address_space(1)))    v4i_t  gv4i;   // global int4
typedef __attribute__((address_space(3)))    v4i_t  lv4i;   // LDS int4

union Frag { v16bf v; uint4 q[2]; };

__device__ __forceinline__ unsigned short f2bf(float f) {
  union { float f; unsigned u; } a; a.f = f;
  unsigned r = a.u + 0x7FFFu + ((a.u >> 16) & 1u);   // round-to-nearest-even
  return (unsigned short)(r >> 16);
}
__device__ __forceinline__ float bf2f(unsigned short h) {
  union { unsigned u; float f; } a; a.u = ((unsigned)h) << 16;
  return a.f;
}

// gfx1250 async global->LDS copy (ASYNCcnt path) with sync fallback
#if defined(__has_builtin)
#if __has_builtin(__builtin_amdgcn_global_load_async_to_lds_b128) && \
    __has_builtin(__builtin_amdgcn_s_wait_asynccnt)
#define HAS_ASYNC_LDS 1
#endif
#endif
#ifndef HAS_ASYNC_LDS
#define HAS_ASYNC_LDS 0
#endif

__device__ __forceinline__ void cp16_to_lds(void* dst_lds, const void* src_global) {
#if HAS_ASYNC_LDS
  gv4i* g = (gv4i*)(unsigned long long)src_global;
  lv4i* l = (lv4i*)(unsigned)(unsigned long long)dst_lds;
  __builtin_amdgcn_global_load_async_to_lds_b128(g, l, 0, 0);
#else
  *(uint4*)dst_lds = *(const uint4*)src_global;
#endif
}
__device__ __forceinline__ void lds_copy_fence() {
#if HAS_ASYNC_LDS
  __builtin_amdgcn_s_wait_asynccnt(0);
#endif
}

// ---------------------------------------------------------------------------
// Kernel 1: fp32 -> bf16 conversion (4-wide)
// ---------------------------------------------------------------------------
__global__ void cvt_f32_bf16_k(const float* __restrict__ s,
                               unsigned short* __restrict__ d, int n4) {
  int i = blockIdx.x * blockDim.x + threadIdx.x;
  int stride = gridDim.x * blockDim.x;
  for (; i < n4; i += stride) {
    float4 v = ((const float4*)s)[i];
    union { unsigned short u[4]; uint2 q; } w;
    w.u[0] = f2bf(v.x); w.u[1] = f2bf(v.y); w.u[2] = f2bf(v.z); w.u[3] = f2bf(v.w);
    ((uint2*)d)[i] = w.q;
  }
}

// ---------------------------------------------------------------------------
// Kernel 2/5: bf16 GEMM  C[M,N] = A[M,K]*B[K,N] + bias, via v_wmma_f32_16x16x32_bf16
// Block: 128x128 tile, BK=32, 256 threads = 8 waves (4x2), wave does 2x4 tiles.
// Double-buffered LDS: stage tile i+1 (async DMA) while WMMAs consume tile i.
// ---------------------------------------------------------------------------
template <bool OUT_BF16>
__global__ __launch_bounds__(256) void gemm_bf16_k(
    const unsigned short* __restrict__ A,   // M x K bf16, row-major
    const unsigned short* __restrict__ B,   // K x N bf16, row-major
    const float* __restrict__ bias,         // N (may be null)
    void* __restrict__ C,                   // M x N (bf16 or f32)
    int M, int N, int K) {
  __shared__ alignas(16) unsigned short As [2][128][48];  // [buf][m][k] (+pad)
  __shared__ alignas(16) unsigned short BsT[2][128][48];  // [buf][n][k] (+pad)

  const int tid = threadIdx.x;
  const int w    = tid >> 5;
  const int lane = tid & 31;
  const int lm   = lane & 15;
  const int hl   = lane >> 4;          // 0: lanes 0-15, 1: lanes 16-31
  const int wm   = w >> 1;             // 0..3  -> rows 32*wm
  const int wn   = w & 1;              // 0..1  -> cols 64*wn
  const int m0   = blockIdx.y * 128;
  const int n0   = blockIdx.x * 128;

  auto stage = [&](int k0, int buf) {
    // A (128x32) row-major: 512 16B chunks, 2 per thread (async path).
#pragma unroll
    for (int it = 0; it < 2; ++it) {
      int c = tid + it * 256;
      int row = c >> 2, ko = (c & 3) * 8;
      cp16_to_lds(&As[buf][row][ko], &A[(size_t)(m0 + row) * K + k0 + ko]);
    }
    // B transposed: BsT[n][k]
#pragma unroll
    for (int it = 0; it < 2; ++it) {
      int c = tid + it * 256;
      int kk = c >> 4, n8 = (c & 15) * 8;
      union { uint4 q; unsigned short s[8]; } u;
      u.q = *(const uint4*)&B[(size_t)(k0 + kk) * N + n0 + n8];
#pragma unroll
      for (int i = 0; i < 8; ++i) BsT[buf][n8 + i][kk] = u.s[i];
    }
  };

  v8f acc[2][4];
#pragma unroll
  for (int mt = 0; mt < 2; ++mt)
#pragma unroll
    for (int nt = 0; nt < 4; ++nt) acc[mt][nt] = (v8f){0,0,0,0,0,0,0,0};

  const int nk = K >> 5;                 // K/32 steps
  stage(0, 0);
  lds_copy_fence();
  __syncthreads();

  for (int i = 0; i < nk; ++i) {
    const int cur = i & 1;
    const bool more = (i + 1) < nk;
    if (more) {
      stage((i + 1) << 5, cur ^ 1);      // overlap DMA with WMMA below
      __builtin_prefetch(&A[(size_t)(m0 + (tid >> 2)) * K + ((i + 2) << 5)], 0, 0);
    }

    // A frag: lanes<16 hold K {0..7,16..23}, lanes>=16 hold {8..15,24..31}
    Frag af[2], bf[4];
#pragma unroll
    for (int mt = 0; mt < 2; ++mt) {
      int row = 32 * wm + 16 * mt + lm;
      af[mt].q[0] = *(const uint4*)&As[cur][row][hl ? 8  : 0];
      af[mt].q[1] = *(const uint4*)&As[cur][row][hl ? 24 : 16];
    }
    // B frag: lanes<16 hold K {0..7,16..23}, lanes>=16 hold {8..15,24..31} (matches A)
#pragma unroll
    for (int nt = 0; nt < 4; ++nt) {
      int col = 64 * wn + 16 * nt + lm;
      bf[nt].q[0] = *(const uint4*)&BsT[cur][col][hl ? 8  : 0];
      bf[nt].q[1] = *(const uint4*)&BsT[cur][col][hl ? 24 : 16];
    }
#pragma unroll
    for (int mt = 0; mt < 2; ++mt)
#pragma unroll
      for (int nt = 0; nt < 4; ++nt)
        acc[mt][nt] = __builtin_amdgcn_wmma_f32_16x16x32_bf16(
            false, af[mt].v, false, bf[nt].v, (short)0, acc[mt][nt], false, false);

    if (more) {
      lds_copy_fence();                  // next buffer's DMA landed
      __syncthreads();                   // all waves done reading cur
    }
  }

  // Epilogue: C layout row = r + 8*hl, col = lane%16 within tile
#pragma unroll
  for (int nt = 0; nt < 4; ++nt) {
    int col = n0 + 64 * wn + 16 * nt + lm;
    float bv = bias ? bias[col] : 0.0f;
#pragma unroll
    for (int mt = 0; mt < 2; ++mt) {
#pragma unroll
      for (int r = 0; r < 8; ++r) {
        int row = m0 + 32 * wm + 16 * mt + r + 8 * hl;
        float val = acc[mt][nt][r] + bv;
        if (OUT_BF16) ((unsigned short*)C)[(size_t)row * N + col] = f2bf(val);
        else          ((float*)C)[(size_t)row * N + col] = val;
      }
    }
  }
}

// ---------------------------------------------------------------------------
// Kernel 3: RoPE (2D axial) + scatter into Q/K/V [B,H,N,64] bf16; Q *= 1/8
// ---------------------------------------------------------------------------
__global__ void rope_scatter_k(const unsigned short* __restrict__ qkv, // [B*N][2304]
                               unsigned short* __restrict__ Q,
                               unsigned short* __restrict__ Kc,
                               unsigned short* __restrict__ V,
                               const int* __restrict__ pos_h,
                               const int* __restrict__ pos_w) {
  const int i  = blockIdx.x * blockDim.x + threadIdx.x;   // over B*N*H*64
  const int d  = i & 63;
  const int hn = i >> 6;
  const int h  = hn % 12;
  const int bn = hn / 12;           // b*1024 + n
  const int b  = bn >> 10;
  const int n  = bn & 1023;

  const unsigned short* base = qkv + (size_t)bn * 2304;
  const int col = h * 64 + d;
  float qv = bf2f(base[col]);
  float kv = bf2f(base[768 + col]);
  float vv = bf2f(base[1536 + col]);

  const int jj = d & 31;                       // index within 32-wide half
  const int pos = (d >> 5) ? pos_w[bn] : pos_h[bn];
  const int f = jj >> 1;                       // frequency index 0..15
  float inv = __powf(10000.0f, -(float)f / 16.0f);
  float p = (float)pos * inv;
  float c = __cosf(p), s = __sinf(p);

  const int partner = (jj < 16) ? (d + 16) : (d - 16);
  const float sgn   = (jj < 16) ? -1.0f : 1.0f;
  float qp = bf2f(base[h * 64 + partner]);
  float kp = bf2f(base[768 + h * 64 + partner]);

  float qo = qv * c + sgn * qp * s;
  float ko = kv * c + sgn * kp * s;

  size_t o = (((size_t)(b * 12 + h)) * 1024 + n) * 64 + d;
  Q[o]  = f2bf(qo * 0.125f);     // fold scale = 64^-0.5
  Kc[o] = f2bf(ko);
  V[o]  = f2bf(vv);
}

// ---------------------------------------------------------------------------
// Kernel 4: flash attention, one WG per (b*h, 128-query block), 8 waves.
// S = Q*K^T (scale folded into Q), online softmax, O = P*V.  AO = [B,N,H,64] bf16
// Double-buffered K/V staging: DMA block j+1 while computing block j.
// ---------------------------------------------------------------------------
__global__ __launch_bounds__(256) void flash_attn_k(
    const unsigned short* __restrict__ Q,
    const unsigned short* __restrict__ Kg,
    const unsigned short* __restrict__ Vg,
    unsigned short* __restrict__ AO) {
  __shared__ alignas(16) unsigned short Ks [2][128][72];   // key row-major (+pad)
  __shared__ alignas(16) unsigned short VsT[2][64][136];   // V transposed: [d][key]
  __shared__ alignas(16) unsigned short Ps [128][136];     // probs, per-wave rows

  const int tid  = threadIdx.x;
  const int w    = tid >> 5;
  const int lane = tid & 31;
  const int lm   = lane & 15;
  const int hl   = lane >> 4;
  const int bh   = blockIdx.y;                 // b*12 + h
  const int b    = bh / 12, h = bh % 12;

  const unsigned short* Qp = Q  + (size_t)bh * 1024 * 64;
  const unsigned short* Kp = Kg + (size_t)bh * 1024 * 64;
  const unsigned short* Vp = Vg + (size_t)bh * 1024 * 64;

  auto stageKV = [&](int j, int buf) {
    // K block row-major (async DMA): 1024 chunks of 16B.
#pragma unroll
    for (int it = 0; it < 4; ++it) {
      int c = tid + it * 256;
      int row = c >> 3, ko = (c & 7) * 8;
      cp16_to_lds(&Ks[buf][row][ko], &Kp[(size_t)(j * 128 + row) * 64 + ko]);
    }
    // V transposed: VsT[d][key]
    for (int idx = tid; idx < 8192; idx += 256) {
      int key = idx >> 6, d = idx & 63;
      VsT[buf][d][key] = Vp[(size_t)(j * 128 + key) * 64 + d];
    }
  };

  // Q fragments (16 rows per wave) held in registers for the whole kernel.
  const int qrow = blockIdx.x * 128 + 16 * w + lm;
  const unsigned short* qptr = Qp + (size_t)qrow * 64;
  Frag qf[2];
#pragma unroll
  for (int ks = 0; ks < 2; ++ks) {
    qf[ks].q[0] = *(const uint4*)(qptr + 32 * ks + (hl ? 8  : 0));
    qf[ks].q[1] = *(const uint4*)(qptr + 32 * ks + (hl ? 24 : 16));
  }

  v8f o[4];
#pragma unroll
  for (int t = 0; t < 4; ++t) o[t] = (v8f){0,0,0,0,0,0,0,0};
  float mrow[8], lrow[8];
#pragma unroll
  for (int r = 0; r < 8; ++r) { mrow[r] = -1e30f; lrow[r] = 0.0f; }

  stageKV(0, 0);
  lds_copy_fence();
  __syncthreads();

  for (int j = 0; j < 8; ++j) {                // key blocks of 128
    const int cur = j & 1;
    const bool more = j < 7;
    if (more) stageKV(j + 1, cur ^ 1);         // overlap DMA with compute below

    // S = Q * K^T : 8 key tiles x 2 K-steps
    v8f s[8];
#pragma unroll
    for (int nt = 0; nt < 8; ++nt) s[nt] = (v8f){0,0,0,0,0,0,0,0};
#pragma unroll
    for (int kk = 0; kk < 2; ++kk) {
#pragma unroll
      for (int nt = 0; nt < 8; ++nt) {
        Frag bfv;
        bfv.q[0] = *(const uint4*)&Ks[cur][16 * nt + lm][32 * kk + (hl ? 8  : 0)];
        bfv.q[1] = *(const uint4*)&Ks[cur][16 * nt + lm][32 * kk + (hl ? 24 : 16)];
        s[nt] = __builtin_amdgcn_wmma_f32_16x16x32_bf16(
            false, qf[kk].v, false, bfv.v, (short)0, s[nt], false, false);
      }
    }

    // Online softmax. Row = r + 8*hl; row data spread over lanes (lm) and nt.
#pragma unroll
    for (int r = 0; r < 8; ++r) {
      float mb = -1e30f;
#pragma unroll
      for (int nt = 0; nt < 8; ++nt) mb = fmaxf(mb, s[nt][r]);
#pragma unroll
      for (int off = 1; off < 16; off <<= 1) mb = fmaxf(mb, __shfl_xor(mb, off, 32));
      float newm  = fmaxf(mrow[r], mb);
      float alpha = __expf(mrow[r] - newm);
      mrow[r] = newm;
      float rs = 0.0f;
#pragma unroll
      for (int nt = 0; nt < 8; ++nt) {
        float pv = __expf(s[nt][r] - newm);
        s[nt][r] = pv;
        rs += pv;
      }
#pragma unroll
      for (int off = 1; off < 16; off <<= 1) rs += __shfl_xor(rs, off, 32);
      lrow[r] = lrow[r] * alpha + rs;
#pragma unroll
      for (int t = 0; t < 4; ++t) o[t][r] *= alpha;
    }

    // C-layout -> A-layout via per-wave LDS region (no barrier needed: same wave).
#pragma unroll
    for (int r = 0; r < 8; ++r)
#pragma unroll
      for (int nt = 0; nt < 8; ++nt)
        Ps[16 * w + r + 8 * hl][16 * nt + lm] = f2bf(s[nt][r]);

    // O += P * V : 4 K-steps (keys) x 4 d-tiles
#pragma unroll
    for (int kk = 0; kk < 4; ++kk) {
      Frag pa;
      pa.q[0] = *(const uint4*)&Ps[16 * w + lm][32 * kk + (hl ? 8  : 0)];
      pa.q[1] = *(const uint4*)&Ps[16 * w + lm][32 * kk + (hl ? 24 : 16)];
#pragma unroll
      for (int t = 0; t < 4; ++t) {
        Frag bv;
        bv.q[0] = *(const uint4*)&VsT[cur][16 * t + lm][32 * kk + (hl ? 8  : 0)];
        bv.q[1] = *(const uint4*)&VsT[cur][16 * t + lm][32 * kk + (hl ? 24 : 16)];
        o[t] = __builtin_amdgcn_wmma_f32_16x16x32_bf16(
            false, pa.v, false, bv.v, (short)0, o[t], false, false);
      }
    }

    if (more) {
      lds_copy_fence();                  // next K block DMA landed
      __syncthreads();                   // all waves done reading cur
    }
  }

  // Normalize and write AO[b][n][h][d]
#pragma unroll
  for (int r = 0; r < 8; ++r) {
    float invl = 1.0f / lrow[r];
    int n_g = blockIdx.x * 128 + 16 * w + r + 8 * hl;
    size_t base = (((size_t)b * 1024 + n_g) * 12 + h) * 64;
#pragma unroll
    for (int t = 0; t < 4; ++t)
      AO[base + 16 * t + lm] = f2bf(o[t][r] * invl);
  }
}

// ---------------------------------------------------------------------------
// Launch
// ---------------------------------------------------------------------------
extern "C" void kernel_launch(void* const* d_in, const int* in_sizes, int n_in,
                              void* d_out, int out_size, void* d_ws, size_t ws_size,
                              hipStream_t stream) {
  (void)in_sizes; (void)n_in; (void)out_size; (void)ws_size;
  const float* x      = (const float*)d_in[0];
  const float* W_qkv  = (const float*)d_in[1];
  const float* b_qkv  = (const float*)d_in[2];
  const float* W_proj = (const float*)d_in[3];
  const float* b_proj = (const float*)d_in[4];
  const int*   pos_h  = (const int*)d_in[5];
  const int*   pos_w  = (const int*)d_in[6];
  float* out = (float*)d_out;

  const int Cc = 768;
  const size_t MN = (size_t)16 * 1024;       // 16384 rows

  // Workspace layout (bf16 regions, 256B-aligned)
  unsigned short* base = (unsigned short*)d_ws;
  size_t off = 0;
  auto alloc = [&](size_t n) {
    unsigned short* r = base + off;
    off += (n + 127) & ~(size_t)127;
    return r;
  };
  unsigned short* xb  = alloc(MN * Cc);                  // x bf16; reused as AO later
  unsigned short* wq  = alloc((size_t)Cc * 3 * Cc);
  unsigned short* wp  = alloc((size_t)Cc * Cc);
  unsigned short* qkv = alloc(MN * 3 * Cc);
  unsigned short* Qb  = alloc(MN * Cc);
  unsigned short* Kb  = alloc(MN * Cc);
  unsigned short* Vb  = alloc(MN * Cc);

  auto cvt = [&](const float* s, unsigned short* d, size_t n) {
    int n4 = (int)(n / 4);
    int blocks = (n4 + 255) / 256;
    if (blocks > 4096) blocks = 4096;
    cvt_f32_bf16_k<<<dim3(blocks), dim3(256), 0, stream>>>(s, d, n4);
  };
  cvt(x,      xb, MN * Cc);
  cvt(W_qkv,  wq, (size_t)Cc * 3 * Cc);
  cvt(W_proj, wp, (size_t)Cc * Cc);

  // qkv = xb @ Wqkv + b_qkv   (bf16 out)
  gemm_bf16_k<true><<<dim3(3 * Cc / 128, (int)(MN / 128)), dim3(256), 0, stream>>>(
      xb, wq, b_qkv, qkv, (int)MN, 3 * Cc, Cc);

  // RoPE + scatter to Q/K/V [B,H,N,64]
  rope_scatter_k<<<dim3((unsigned)(MN * 12 * 64 / 256)), dim3(256), 0, stream>>>(
      qkv, Qb, Kb, Vb, pos_h, pos_w);

  // Flash attention -> AO (reuse xb region, same element count)
  flash_attn_k<<<dim3(1024 / 128, 16 * 12), dim3(256), 0, stream>>>(Qb, Kb, Vb, xb);

  // out = AO @ Wproj + b_proj   (f32 out)
  gemm_bf16_k<false><<<dim3(Cc / 128, (int)(MN / 128)), dim3(256), 0, stream>>>(
      xb, wp, b_proj, out, (int)MN, Cc, Cc);
}
